// Head_8572754722898
// MI455X (gfx1250) — matrix operand order repaired and checked by reading, not compiled, not run on hardware
//
#include <hip/hip_runtime.h>

typedef __attribute__((ext_vector_type(16))) __bf16 v16bf;
typedef __attribute__((ext_vector_type(8)))  float  v8f;

#define B_N 16
#define T_N 2048
#define C_N 1024
#define H_N 128
#define M_ROWS (B_N * T_N)   // 32768

// Flip to 0 if the assembler rejects the async-to-LDS mnemonics.
#define USE_ASYNC_COPY 1

__device__ __forceinline__ v8f wmma_bf16f32(v16bf a, v16bf b, v8f c) {
  // D = A(16x32 bf16) * B(32x16 bf16) + C(16x16 f32)
  return __builtin_amdgcn_wmma_f32_16x16x32_bf16(
      false, a, false, b, (short)0, c, false, false);
}

// 16-byte global -> LDS copy. Async path uses the CDNA5 ASYNCcnt-tracked DMA
// (generic LDS pointer low 32 bits == LDS byte address per the aperture rules).
__device__ __forceinline__ void copy_b128(void* lds, const void* g) {
#if USE_ASYNC_COPY
  asm volatile("global_load_async_to_lds_b128 %0, %1, off"
               :: "v"((unsigned)(unsigned long long)lds),
                  "v"((unsigned long long)g)
               : "memory");
#else
  *(uint4*)lds = *(const uint4*)g;
#endif
}

__device__ __forceinline__ void async_wait() {
#if USE_ASYNC_COPY
  asm volatile("s_wait_asynccnt 0" ::: "memory");
#endif
}

// ---------------------------------------------------------------------------
// Pass 1: fused q/k/v = x @ {Wq,Wk,Wv}. x read ONCE (134 MB, ~6us at 23.3TB/s).
// 512 threads = 16 waves: wave (rowtile = w&7, colhalf = w>>3) computes
// 16 rows x 64 cols of all three projections -> 3x4 accumulator groups
// (96 VGPRs/wave, keeps allocation < 256, no s_set_vgpr_msb churn).
// Software-pipelined: chunk kc+1's float4 loads are in flight during chunk
// kc's 12 WMMAs. v is stored TRANSPOSED ([H][M]); per lane the 8 accumulator
// rows are consecutive M, so each 16x16 group packs into one b128 store.
// ---------------------------------------------------------------------------
__global__ __launch_bounds__(512) void qkv_project_kernel(
    const float* __restrict__ x,
    const float* __restrict__ Wq,
    const float* __restrict__ Wk,
    const float* __restrict__ Wv,
    __bf16* __restrict__ qb,     // [M][H] row-major bf16
    __bf16* __restrict__ kb,     // [M][H] row-major bf16
    __bf16* __restrict__ vbT)    // [H][M] transposed bf16
{
  __shared__ __bf16 xs[128 * 32];        // [row][k]        8 KB
  __shared__ __bf16 wtT[3 * 128 * 32];   // [p][n][k]      24 KB (W transposed)

  const int tid     = threadIdx.x;
  const int wave    = tid >> 5;
  const int lane    = tid & 31;
  const int lanelow = lane & 15;
  const int half    = lane >> 4;
  const int rowtile = wave & 7;
  const int colhalf = wave >> 3;         // 0: cols 0-63, 1: cols 64-127
  const int m0      = blockIdx.x * 128;

  v8f acc[3][4];
  #pragma unroll
  for (int p = 0; p < 3; ++p)
    #pragma unroll
    for (int g = 0; g < 4; ++g)
      #pragma unroll
      for (int r = 0; r < 8; ++r) acc[p][g][r] = 0.0f;

  float4 xf[2];       // prefetch registers: x tile (2 float4/thread)
  float4 wf[3][2];    // prefetch registers: W tiles (6 float4/thread)

  auto load_regs = [&](int kc) {
    #pragma unroll
    for (int it = 0; it < 2; ++it) {
      int qi  = tid + it * 512;          // float4 index 0..1023
      int row = qi >> 3;
      int col = (qi & 7) * 4;
      xf[it] = *(const float4*)&x[(size_t)(m0 + row) * C_N + kc + col];
    }
    #pragma unroll
    for (int p = 0; p < 3; ++p) {
      const float* W = (p == 0) ? Wq : ((p == 1) ? Wk : Wv);
      #pragma unroll
      for (int it = 0; it < 2; ++it) {
        int qi  = tid + it * 512;
        int row = qi >> 5;               // k-in-tile 0..31
        int col = (qi & 31) * 4;         // n
        wf[p][it] = *(const float4*)&W[(size_t)(kc + row) * H_N + col];
      }
    }
  };

  auto store_lds = [&]() {
    #pragma unroll
    for (int it = 0; it < 2; ++it) {
      int qi  = tid + it * 512;
      int row = qi >> 3;
      int col = (qi & 7) * 4;
      xs[row * 32 + col + 0] = (__bf16)xf[it].x;
      xs[row * 32 + col + 1] = (__bf16)xf[it].y;
      xs[row * 32 + col + 2] = (__bf16)xf[it].z;
      xs[row * 32 + col + 3] = (__bf16)xf[it].w;
    }
    #pragma unroll
    for (int p = 0; p < 3; ++p) {
      __bf16* wt = wtT + p * (128 * 32);
      #pragma unroll
      for (int it = 0; it < 2; ++it) {
        int qi  = tid + it * 512;
        int row = qi >> 5;
        int col = (qi & 31) * 4;
        wt[(col + 0) * 32 + row] = (__bf16)wf[p][it].x;
        wt[(col + 1) * 32 + row] = (__bf16)wf[p][it].y;
        wt[(col + 2) * 32 + row] = (__bf16)wf[p][it].z;
        wt[(col + 3) * 32 + row] = (__bf16)wf[p][it].w;
      }
    }
  };

  load_regs(0);
  for (int kc = 0; kc < C_N; kc += 32) {
    store_lds();
    __syncthreads();
    if (kc + 32 < C_N) load_regs(kc + 32);   // overlap HBM with the WMMAs below

    // One A fragment feeds 12 WMMAs
    v16bf a;
    {
      const __bf16* pa = &xs[(rowtile * 16 + lanelow) * 32 + half * 16];
      #pragma unroll
      for (int e = 0; e < 8; ++e) a[e] = pa[e];
      #pragma unroll
      for (int e = 0; e < 8; ++e) a[8 + e] = pa[8 + e];
    }
    #pragma unroll
    for (int p = 0; p < 3; ++p)
      #pragma unroll
      for (int g = 0; g < 4; ++g) {
        int gg = colhalf * 4 + g;
        v16bf bf;
        const __bf16* pb = &wtT[p * (128 * 32) + (gg * 16 + lanelow) * 32 + half * 16];
        #pragma unroll
        for (int e = 0; e < 16; ++e) bf[e] = pb[e];
        acc[p][g] = wmma_bf16f32(a, bf, acc[p][g]);
      }
    __syncthreads();
  }

  // q, k: row-major b16 stores (lanes 0-15 cover 16 consecutive columns)
  #pragma unroll
  for (int g = 0; g < 4; ++g)
    #pragma unroll
    for (int r = 0; r < 8; ++r) {
      int row = m0 + rowtile * 16 + r + 8 * half;
      int col = (colhalf * 4 + g) * 16 + lanelow;
      qb[(size_t)row * H_N + col] = (__bf16)acc[0][g][r];
      kb[(size_t)row * H_N + col] = (__bf16)acc[1][g][r];
    }
  // v: transposed layout; 8 consecutive rows per lane pack into one b128 store
  #pragma unroll
  for (int g = 0; g < 4; ++g) {
    unsigned w4[4];
    #pragma unroll
    for (int i = 0; i < 4; ++i) {
      unsigned short lo = __builtin_bit_cast(unsigned short, (__bf16)acc[2][g][2 * i]);
      unsigned short hi = __builtin_bit_cast(unsigned short, (__bf16)acc[2][g][2 * i + 1]);
      w4[i] = (unsigned)lo | ((unsigned)hi << 16);
    }
    int col = (colhalf * 4 + g) * 16 + lanelow;
    size_t off = (size_t)col * M_ROWS + (size_t)(m0 + rowtile * 16 + 8 * half);
    uint4 val; val.x = w4[0]; val.y = w4[1]; val.z = w4[2]; val.w = w4[3];
    *(uint4*)&vbT[off] = val;
  }
}

// ---------------------------------------------------------------------------
// Pass 2: causal flash attention. Block = (qt, batch), 256 threads (8 waves).
// Double-buffered K/V tiles staged with async-to-LDS DMA; tile kt+1 streams in
// while the 64 WMMAs of tile kt execute. Softmax scale is folded into the
// exp2 argument (max commutes with positive scaling).
// ---------------------------------------------------------------------------
__global__ __launch_bounds__(256) void flash_attn_kernel(
    const __bf16* __restrict__ qb,
    const __bf16* __restrict__ kb,
    const __bf16* __restrict__ vbT,
    float* __restrict__ out)
{
  extern __shared__ __bf16 smem[];
  // NOTE: no pointer arrays over LDS (clang emits addrspacecast global
  // initializers that lld rejects) -- scalar pointers + ternary selects only.
  __bf16* ks0 = smem;                    // [key][h] buffer 0
  __bf16* vt0 = smem + 16384;            // [h][key] buffer 0
  __bf16* ks1 = smem + 2 * 16384;        // [key][h] buffer 1
  __bf16* vt1 = smem + 3 * 16384;        // [h][key] buffer 1
  __bf16* ps  = smem + 4 * 16384;        // per-wave P staging [16][128] (32 KB)

  const int tid     = threadIdx.x;
  const int wave    = tid >> 5;
  const int lane    = tid & 31;
  const int lanelow = lane & 15;
  const int half    = lane >> 4;

  const int qt    = blockIdx.x;
  const int b     = blockIdx.y;
  const int qbase = b * T_N + qt * 128;

  // Q fragments (4 K-chunks of 32 over H=128), kept in registers
  v16bf aq[4];
  {
    const __bf16* qp = &qb[(size_t)(qbase + wave * 16 + lanelow) * H_N];
    #pragma unroll
    for (int kc = 0; kc < 4; ++kc) {
      const __bf16* p = qp + kc * 32 + half * 16;
      #pragma unroll
      for (int e = 0; e < 8; ++e) aq[kc][e] = p[e];
      #pragma unroll
      for (int e = 0; e < 8; ++e) aq[kc][8 + e] = p[8 + e];
    }
  }

  v8f c[8];
  float mrow[8], lrow[8];
  #pragma unroll
  for (int r = 0; r < 8; ++r) { mrow[r] = -3.0e38f; lrow[r] = 0.0f; }
  #pragma unroll
  for (int g = 0; g < 8; ++g)
    #pragma unroll
    for (int r = 0; r < 8; ++r) c[g][r] = 0.0f;

  // exp(s*scale - m*scale) == exp2((s - m) * sl), sl = log2(e)/sqrt(H)
  const float sl = 1.4426950408889634f * 0.08838834764831845f;

  // Stage key tile kt2 into buffer (K straight copy; V rows from transposed vbT)
  auto stage = [&](int kt2, __bf16* ksd, __bf16* vtd) {
    const size_t krow = (size_t)b * T_N + (size_t)kt2 * 128;
    const uint4* ksrc = (const uint4*)(kb + krow * H_N);
    uint4* kdst = (uint4*)ksd;
    uint4* vdst = (uint4*)vtd;
    #pragma unroll
    for (int it = 0; it < 8; ++it) {
      int q = tid + it * 256;
      copy_b128(&kdst[q], &ksrc[q]);
    }
    #pragma unroll
    for (int it = 0; it < 8; ++it) {
      int q   = tid + it * 256;
      int h   = q >> 4;
      int c16 = q & 15;
      const uint4* vsrc = (const uint4*)(vbT + (size_t)h * M_ROWS + krow);
      copy_b128(&vdst[q], &vsrc[c16]);
    }
  };

  stage(0, ks0, vt0);
  async_wait();
  __syncthreads();

  for (int kt = 0; kt <= qt; ++kt) {
    const int cur = kt & 1;
    const __bf16* ks = cur ? ks1 : ks0;
    const __bf16* vt = cur ? vt1 : vt0;
    if (kt < qt)  // overlap next tile's DMA with this tile's 64 WMMAs
      stage(kt + 1, cur ? ks0 : ks1, cur ? vt0 : vt1);

    // S = Q * K^T : 8 key groups x 4 K-chunks = 32 WMMAs (raw, unscaled scores)
    v8f s[8];
    #pragma unroll
    for (int g = 0; g < 8; ++g) {
      v8f sg;
      #pragma unroll
      for (int r = 0; r < 8; ++r) sg[r] = 0.0f;
      #pragma unroll
      for (int kc = 0; kc < 4; ++kc) {
        v16bf bf;
        const __bf16* pb = &ks[(g * 16 + lanelow) * 128 + kc * 32 + half * 16];
        #pragma unroll
        for (int e = 0; e < 16; ++e) bf[e] = pb[e];
        sg = wmma_bf16f32(aq[kc], bf, sg);
      }
      s[g] = sg;
    }

    // causal mask only on the diagonal tile (raw-score domain)
    if (kt == qt) {
      #pragma unroll
      for (int g = 0; g < 8; ++g) {
        int kl = g * 16 + lanelow;
        #pragma unroll
        for (int r = 0; r < 8; ++r) {
          int ql = wave * 16 + r + 8 * half;
          s[g][r] = (kl > ql) ? -3.0e38f : s[g][r];
        }
      }
    }

    // online softmax: row reductions across the 16-lane half (xor 1,2,4,8)
    #pragma unroll
    for (int r = 0; r < 8; ++r) {
      float mx = -3.0e38f;
      #pragma unroll
      for (int g = 0; g < 8; ++g) mx = fmaxf(mx, s[g][r]);
      #pragma unroll
      for (int d = 1; d < 16; d <<= 1) mx = fmaxf(mx, __shfl_xor(mx, d, 32));
      float mn = fmaxf(mrow[r], mx);
      float al = __builtin_exp2f((mrow[r] - mn) * sl);
      mrow[r] = mn;
      float nb = -mn * sl;
      float rs = 0.0f;
      #pragma unroll
      for (int g = 0; g < 8; ++g) {
        float p = __builtin_exp2f(fmaf(s[g][r], sl, nb));
        s[g][r] = p;
        rs += p;
      }
      #pragma unroll
      for (int d = 1; d < 16; d <<= 1) rs += __shfl_xor(rs, d, 32);
      lrow[r] = lrow[r] * al + rs;
      #pragma unroll
      for (int g = 0; g < 8; ++g) c[g][r] *= al;
    }

    // P (C-layout) -> per-wave LDS -> A-layout fragments
    __bf16* pw = ps + wave * 16 * 128;
    #pragma unroll
    for (int g = 0; g < 8; ++g)
      #pragma unroll
      for (int r = 0; r < 8; ++r)
        pw[(r + 8 * half) * 128 + g * 16 + lanelow] = (__bf16)s[g][r];

    v16bf ap[4];
    #pragma unroll
    for (int kc = 0; kc < 4; ++kc) {
      const __bf16* p = &pw[lanelow * 128 + kc * 32 + half * 16];
      #pragma unroll
      for (int e = 0; e < 8; ++e) ap[kc][e] = p[e];
      #pragma unroll
      for (int e = 0; e < 8; ++e) ap[kc][8 + e] = p[8 + e];
    }

    // O += P * V : 8 H-groups x 4 key-chunks = 32 WMMAs
    #pragma unroll
    for (int g = 0; g < 8; ++g)
      #pragma unroll
      for (int kc = 0; kc < 4; ++kc) {
        v16bf bf;
        const __bf16* pb = &vt[(g * 16 + lanelow) * 128 + kc * 32 + half * 16];
        #pragma unroll
        for (int e = 0; e < 16; ++e) bf[e] = pb[e];
        c[g] = wmma_bf16f32(ap[kc], bf, c[g]);
      }

    async_wait();       // next tile's DMA must have landed before the barrier
    __syncthreads();
  }

  // normalize and write f32 output [B,T,H]
  float inv[8];
  #pragma unroll
  for (int r = 0; r < 8; ++r) inv[r] = 1.0f / lrow[r];
  #pragma unroll
  for (int g = 0; g < 8; ++g)
    #pragma unroll
    for (int r = 0; r < 8; ++r) {
      int row = qbase + wave * 16 + r + 8 * half;
      int col = g * 16 + lanelow;
      out[(size_t)row * H_N + col] = c[g][r] * inv[r];
    }
}

// ---------------------------------------------------------------------------
extern "C" void kernel_launch(void* const* d_in, const int* in_sizes, int n_in,
                              void* d_out, int out_size, void* d_ws, size_t ws_size,
                              hipStream_t stream) {
  const float* x  = (const float*)d_in[0];
  const float* Wq = (const float*)d_in[1];
  const float* Wk = (const float*)d_in[2];
  const float* Wv = (const float*)d_in[3];

  __bf16* qb  = (__bf16*)d_ws;                      // [M][H] bf16
  __bf16* kb  = qb + (size_t)M_ROWS * H_N;          // [M][H] bf16
  __bf16* vbT = kb + (size_t)M_ROWS * H_N;          // [H][M] bf16

  qkv_project_kernel<<<dim3(M_ROWS / 128), dim3(512), 0, stream>>>(
      x, Wq, Wk, Wv, qb, kb, vbT);

  dim3 g2(T_N / 128, B_N);
  size_t shmem = (size_t)(4 * 16384 + 8 * 16 * 128) * sizeof(__bf16); // 160 KB
  flash_attn_kernel<<<g2, dim3(256), shmem, stream>>>(qb, kb, vbT, (float*)d_out);
}
